// ISTFT_46712064311790
// MI455X (gfx1250) — matrix operand order repaired and checked by reading, not compiled, not run on hardware
//
#include <hip/hip_runtime.h>

typedef __attribute__((ext_vector_type(16))) _Float16 v16h;
typedef __attribute__((ext_vector_type(8)))  float    v8f;

#define NFFT    1024
#define HOP     256
#define PADT    384
#define FBINS   513
#define TFR     2048
#define NBATCH  16
#define KTILES  33          // K padded: 1026 -> 1056 = 33 * 32
#define MTILES  64          // 1024 rows / 16
#define OUTFULL 525056      // (2048-1)*256 + 1024
#define OUTTRIM 524288      // OUTFULL - 2*PADT
#define BSTRIDE 40          // LDS halves per column (16B aligned, bank-spread)

// ---------------------------------------------------------------------------
// Build windowed inverse-rDFT basis, pre-packed into the CDNA5 WMMA A-matrix
// per-lane order used here (contiguous k, same as the B reads): wp[((mt*33 + kt)*32 + lane)*16 + e]
// lane = (half<<4)|m_low holds row m = mt*16 + m_low,
//   K = kt*32 + 16*half + e
// Column K < 513: real-part coeff; 513 <= K < 1026: imag-part; else 0 (pad).
// ---------------------------------------------------------------------------
__global__ void build_w_kernel(const float* __restrict__ window,
                               _Float16* __restrict__ wp) {
  int idx  = blockIdx.x * 256 + threadIdx.x;
  int e    = idx & 15;
  int lane = (idx >> 4) & 31;
  int rest = idx >> 9;
  int kt   = rest % KTILES;
  int mt   = rest / KTILES;
  if (mt >= MTILES) return;
  int m    = mt * 16 + (lane & 15);
  int half = lane >> 4;
  int kk   = 16 * half + e;
  int K    = kt * 32 + kk;
  float coef = 0.0f;
  if (K < 2 * FBINS) {
    int   k   = (K < FBINS) ? K : (K - FBINS);
    int   ph  = (k * m) & (NFFT - 1);                 // exact phase reduction
    float ang = (float)ph * (6.283185307179586f / (float)NFFT);
    float w   = window[m] * (1.0f / (float)NFFT);
    if (K < FBINS) {
      float a = (k == 0 || k == NFFT / 2) ? 1.0f : 2.0f;
      coef = a * cosf(ang) * w;
    } else if (k != 0 && k != NFFT / 2) {             // imag of DC/Nyquist ignored
      coef = -2.0f * sinf(ang) * w;
    }
  }
  wp[idx] = (_Float16)coef;
}

// ---------------------------------------------------------------------------
// Per-batch GEMM D[1024 x 2048] = W[1024 x 1056] * S[1056 x 2048], fused with
// windowing (in W) and overlap-add scatter (atomic f32 into accum plane).
// Block: 256 threads (8 waves). Grid: (T/64, 1024/256, batch).
// Wave w: M-tiles {2w, 2w+1} of the 256-row block, all 4 N-tiles.
// Main loop (kt 0..31) is branch-free: row source is a pointer select.
// kt=32 (zero-padded tail rows 1024..1055) is peeled.
// WMMAs run in two passes over bt[0..3] so all 4 B tiles must stay live ->
// scheduler hoists all 8 ds_loads above one wait, then bursts 8 WMMAs.
// ---------------------------------------------------------------------------
__global__ __launch_bounds__(256) void istft_gemm_kernel(
    const float* __restrict__ sre, const float* __restrict__ sim,
    const _Float16* __restrict__ wp, float* __restrict__ accum) {
  __shared__ __align__(16) _Float16 bsh[2][64 * BSTRIDE];

  const int tid  = threadIdx.x;
  const int lane = tid & 31;
  const int wave = tid >> 5;
  const int t0   = blockIdx.x * 64;
  const int m0   = blockIdx.y * 256;
  const int b    = blockIdx.z;

  const float* __restrict__ srb = sre + (size_t)b * FBINS * TFR;
  const float* __restrict__ sib = sim + (size_t)b * FBINS * TFR;

  // B-staging assignment: 32 K-rows x 64 t; each thread: 1 row, 8 contiguous t
  const int lkk = tid >> 3;        // 0..31  (row within K-step)
  const int lt  = (tid & 7) * 8;   // 0,8,..,56 (t offset)

  v8f acc[2][4];
  const v8f vzero = {};
  #pragma unroll
  for (int i = 0; i < 2; ++i)
    #pragma unroll
    for (int j = 0; j < 4; ++j) acc[i][j] = vzero;

  // A (packed W) base indices for this wave's two M tiles
  const int mt0 = (m0 >> 4) + wave * 2;

  // B read offset: lane n reads column nt*16+n, halves [16*half, 16*half+16)
  const int boff = (lane & 15) * BSTRIDE + (lane >> 4) * 16;
  const int lds_wr = lt * BSTRIDE + lkk;

  #pragma unroll 1
  for (int kt = 0; kt < KTILES - 1; ++kt) {
    const int buf = kt & 1;
    // ---- stage B tile (32 x 64) f32 -> f16 into LDS, branch-free ----
    const int r = kt * 32 + lkk;                     // always < 1024 here
    const float* src = (r < FBINS)
        ? (srb + (size_t)r * TFR + t0 + lt)          // real rows 0..512
        : (sib + (size_t)(r - FBINS) * TFR + t0 + lt);  // imag rows 513..1023
    const float4 f0 = ((const float4*)src)[0];
    const float4 f1 = ((const float4*)src)[1];
    const _Float16 h[8] = {(_Float16)f0.x, (_Float16)f0.y, (_Float16)f0.z, (_Float16)f0.w,
                           (_Float16)f1.x, (_Float16)f1.y, (_Float16)f1.z, (_Float16)f1.w};
    #pragma unroll
    for (int j = 0; j < 8; ++j)
      bsh[buf][lds_wr + j * BSTRIDE] = h[j];

    // ---- A loads issued BEFORE the barrier: latency hidden by barrier wait ----
    const v16h a0 = *(const v16h*)(wp + ((size_t)((mt0 + 0) * KTILES + kt) * 32 + lane) * 16);
    const v16h a1 = *(const v16h*)(wp + ((size_t)((mt0 + 1) * KTILES + kt) * 32 + lane) * 16);

    __syncthreads();

    v16h bt0 = *(const v16h*)(&bsh[buf][0 * 16 * BSTRIDE + boff]);
    v16h bt1 = *(const v16h*)(&bsh[buf][1 * 16 * BSTRIDE + boff]);
    v16h bt2 = *(const v16h*)(&bsh[buf][2 * 16 * BSTRIDE + boff]);
    v16h bt3 = *(const v16h*)(&bsh[buf][3 * 16 * BSTRIDE + boff]);

    // pass 1: a0 against all 4 tiles (keeps bt0..bt3 live), pass 2: a1
    acc[0][0] = __builtin_amdgcn_wmma_f32_16x16x32_f16(false, a0, false, bt0, (short)0, acc[0][0], false, false);
    acc[0][1] = __builtin_amdgcn_wmma_f32_16x16x32_f16(false, a0, false, bt1, (short)0, acc[0][1], false, false);
    acc[0][2] = __builtin_amdgcn_wmma_f32_16x16x32_f16(false, a0, false, bt2, (short)0, acc[0][2], false, false);
    acc[0][3] = __builtin_amdgcn_wmma_f32_16x16x32_f16(false, a0, false, bt3, (short)0, acc[0][3], false, false);
    acc[1][0] = __builtin_amdgcn_wmma_f32_16x16x32_f16(false, a1, false, bt0, (short)0, acc[1][0], false, false);
    acc[1][1] = __builtin_amdgcn_wmma_f32_16x16x32_f16(false, a1, false, bt1, (short)0, acc[1][1], false, false);
    acc[1][2] = __builtin_amdgcn_wmma_f32_16x16x32_f16(false, a1, false, bt2, (short)0, acc[1][2], false, false);
    acc[1][3] = __builtin_amdgcn_wmma_f32_16x16x32_f16(false, a1, false, bt3, (short)0, acc[1][3], false, false);
  }

  // ---- peeled tail: kt = 32, rows 1024..1055 (imag 511,512 then zero pad) ----
  {
    const int kt  = KTILES - 1;
    const int buf = kt & 1;   // = 0
    const int r   = kt * 32 + lkk;
    float4 f0{}, f1{};
    if (r < 2 * FBINS) {      // only r = 1024, 1025 (imag k = 511, 512)
      const float4* p = (const float4*)(sib + (size_t)(r - FBINS) * TFR + t0 + lt);
      f0 = p[0]; f1 = p[1];
    }
    const _Float16 h[8] = {(_Float16)f0.x, (_Float16)f0.y, (_Float16)f0.z, (_Float16)f0.w,
                           (_Float16)f1.x, (_Float16)f1.y, (_Float16)f1.z, (_Float16)f1.w};
    #pragma unroll
    for (int j = 0; j < 8; ++j)
      bsh[buf][lds_wr + j * BSTRIDE] = h[j];

    const v16h a0 = *(const v16h*)(wp + ((size_t)((mt0 + 0) * KTILES + kt) * 32 + lane) * 16);
    const v16h a1 = *(const v16h*)(wp + ((size_t)((mt0 + 1) * KTILES + kt) * 32 + lane) * 16);

    __syncthreads();

    v16h bt0 = *(const v16h*)(&bsh[buf][0 * 16 * BSTRIDE + boff]);
    v16h bt1 = *(const v16h*)(&bsh[buf][1 * 16 * BSTRIDE + boff]);
    v16h bt2 = *(const v16h*)(&bsh[buf][2 * 16 * BSTRIDE + boff]);
    v16h bt3 = *(const v16h*)(&bsh[buf][3 * 16 * BSTRIDE + boff]);

    acc[0][0] = __builtin_amdgcn_wmma_f32_16x16x32_f16(false, a0, false, bt0, (short)0, acc[0][0], false, false);
    acc[0][1] = __builtin_amdgcn_wmma_f32_16x16x32_f16(false, a0, false, bt1, (short)0, acc[0][1], false, false);
    acc[0][2] = __builtin_amdgcn_wmma_f32_16x16x32_f16(false, a0, false, bt2, (short)0, acc[0][2], false, false);
    acc[0][3] = __builtin_amdgcn_wmma_f32_16x16x32_f16(false, a0, false, bt3, (short)0, acc[0][3], false, false);
    acc[1][0] = __builtin_amdgcn_wmma_f32_16x16x32_f16(false, a1, false, bt0, (short)0, acc[1][0], false, false);
    acc[1][1] = __builtin_amdgcn_wmma_f32_16x16x32_f16(false, a1, false, bt1, (short)0, acc[1][1], false, false);
    acc[1][2] = __builtin_amdgcn_wmma_f32_16x16x32_f16(false, a1, false, bt2, (short)0, acc[1][2], false, false);
    acc[1][3] = __builtin_amdgcn_wmma_f32_16x16x32_f16(false, a1, false, bt3, (short)0, acc[1][3], false, false);
  }

  // ---- overlap-add scatter: p = t*HOP + n; all p distinct inside this WG ----
  float* __restrict__ ab = accum + (size_t)b * OUTFULL;
  #pragma unroll
  for (int i = 0; i < 2; ++i) {
    const int nbase = m0 + wave * 32 + i * 16 + (lane >> 4) * 8;  // row M base
    #pragma unroll
    for (int nt = 0; nt < 4; ++nt) {
      const int t = t0 + nt * 16 + (lane & 15);
      float* base = ab + (size_t)t * HOP + nbase;
      #pragma unroll
      for (int v = 0; v < 8; ++v)
        unsafeAtomicAdd(base + v, acc[i][nt][v]);
    }
  }
}

// ---------------------------------------------------------------------------
// Envelope normalization + trim. env[p] = sum_j window[p%256 + 256j]^2 over
// valid frames t = p/256 - j; constant-time (4 taps), no precomputed buffer.
// ---------------------------------------------------------------------------
__global__ void finalize_kernel(const float* __restrict__ accum,
                                const float* __restrict__ window,
                                float* __restrict__ out) {
  int idx = blockIdx.x * 256 + threadIdx.x;
  if (idx >= NBATCH * OUTTRIM) return;
  int b  = idx / OUTTRIM;
  int p  = (idx % OUTTRIM) + PADT;
  int nb = p & (HOP - 1);
  int tb = p >> 8;
  float env = 0.0f;
  #pragma unroll
  for (int j = 0; j < 4; ++j) {
    int t = tb - j;
    if (t >= 0 && t < TFR) {
      float w = window[nb + HOP * j];
      env += w * w;
    }
  }
  env = fmaxf(env, 1e-11f);
  out[idx] = accum[(size_t)b * OUTFULL + p] / env;
}

// ---------------------------------------------------------------------------
// Workspace layout: [0, 4MB) packed-f16 W basis (2.16MB used);
//                   [4MB, 4MB + 16*525056*4) f32 overlap-add accumulator.
// Total ~37.6 MB.
// ---------------------------------------------------------------------------
extern "C" void kernel_launch(void* const* d_in, const int* in_sizes, int n_in,
                              void* d_out, int out_size, void* d_ws, size_t ws_size,
                              hipStream_t stream) {
  const float* sre    = (const float*)d_in[0];
  const float* sim    = (const float*)d_in[1];
  const float* window = (const float*)d_in[2];
  float*       out    = (float*)d_out;

  char*     ws    = (char*)d_ws;
  _Float16* wp    = (_Float16*)ws;
  float*    accum = (float*)(ws + (size_t)(4 << 20));

  hipMemsetAsync(accum, 0, (size_t)NBATCH * OUTFULL * sizeof(float), stream);

  const int wtotal = MTILES * KTILES * 32 * 16;  // 1,081,344 halves
  build_w_kernel<<<(wtotal + 255) / 256, 256, 0, stream>>>(window, wp);

  dim3 grid(TFR / 64, 1024 / 256, NBATCH);
  istft_gemm_kernel<<<grid, 256, 0, stream>>>(sre, sim, wp, accum);

  finalize_kernel<<<(NBATCH * OUTTRIM) / 256, 256, 0, stream>>>(accum, window, out);
}
